// GBFormer_10720238371007
// MI455X (gfx1250) — matrix operand order repaired and checked by reading, not compiled, not run on hardware
//
#include <hip/hip_runtime.h>
#include <math.h>

// ---------------------------------------------------------------------------
// GBFormer for MI455X (gfx1250).  All GEMM-shaped einsums are lowered to
// v_wmma_f32_16x16x32_f16 (f16 inputs, f32 accumulate).  Activations fit in
// the 192MB L2 so the workload is matrix-op bound -> WMMA everywhere.
// Weight (B) tiles are DMA'd into LDS with the Tensor Data Mover
// (tensor_load_to_lds + s_wait_tensorcnt) and shared by all 4 waves of a
// block; A streams from global with global_prefetch hints.
// Reductions / exp / sort stay fp32 VALU.
// ---------------------------------------------------------------------------

#define B_      2
#define G_      19200
#define E_      256
#define NTOK    (B_ * G_)          // 38400 tokens
#define BINS_   10
#define LBIN    (G_ / BINS_)       // 1920
#define HEADS_  4
#define DH_     64
#define MFEAT_  256
#define DFF_    1024
#define NEDGES_ 153600
#define SORTN   32768
#define SMAX    (B_ * BINS_)       // 20 sequences max

#define LN_EPS_ 1e-5f
#define KEPS_   1e-4f
#define DN_     0.35355339059327373f   // DH^-0.25
#define RATIO_  0.0625f                // MFEAT^-0.5

typedef __attribute__((ext_vector_type(16))) _Float16 v16h;
typedef __attribute__((ext_vector_type(8)))  float    v8f;
typedef __attribute__((ext_vector_type(4)))  unsigned v4u;
typedef __attribute__((ext_vector_type(8)))  int      v8i;
typedef __attribute__((ext_vector_type(4)))  int      v4i;

#ifndef __has_builtin
#define __has_builtin(x) 0
#endif
#if __has_builtin(__builtin_amdgcn_tensor_load_to_lds) && \
    __has_builtin(__builtin_amdgcn_s_wait_tensorcnt)
#define HAVE_TDM 1
#else
#define HAVE_TDM 0
#endif

#if HAVE_TDM
// 2D tile DMA: global [rows x kElems] f16 (row stride = strideElems) -> LDS,
// packed contiguous [rows][kElems].  D# built per CDNA5 ISA.
__device__ inline void tdm_load_b_tile(unsigned lds_off, const _Float16* gsrc,
                                       int kElems, int rows, long long strideElems) {
    unsigned long long ga = (unsigned long long)(const void*)gsrc;
    v4u g0;
    g0.x = 1u;                                             // count=1 (user D#)
    g0.y = lds_off;                                        // lds_addr (bytes)
    g0.z = (unsigned)(ga & 0xffffffffull);                 // global_addr[31:0]
    g0.w = (unsigned)((ga >> 32) & 0x01ffffffull) | (2u << 30);  // addr[56:32], type=2
    v8i g1;
    g1[0] = 0x00010000;                                    // data_size = 2 bytes
    g1[1] = (int)((unsigned)(kElems & 0xffff) << 16);      // tensor_dim0[15:0]
    g1[2] = (int)(((unsigned)(kElems >> 16) & 0xffffu) |
                  ((unsigned)(rows & 0xffff) << 16));      // dim0 hi | tensor_dim1 lo
    g1[3] = (int)(((unsigned)(rows >> 16) & 0xffffu) |
                  ((unsigned)(kElems & 0xffff) << 16));    // dim1 hi | tile_dim0
    g1[4] = (int)(rows & 0xffff);                          // tile_dim1 (tile_dim2=0)
    g1[5] = (int)(strideElems & 0xffffffffll);             // tensor_dim0_stride lo
    g1[6] = (int)((strideElems >> 32) & 0xffffll);         // stride hi (dim1_stride=0)
    g1[7] = 0;
    v4i z4 = {0, 0, 0, 0};
#if __clang_major__ >= 23
    v8i z8 = {0, 0, 0, 0, 0, 0, 0, 0};
    __builtin_amdgcn_tensor_load_to_lds(g0, g1, z4, z4, z8, 0);
#else
    __builtin_amdgcn_tensor_load_to_lds(g0, g1, z4, z4, 0);
#endif
}
#endif

// ------------------------------ reductions ---------------------------------
__device__ inline float block_reduce_sum256(float v) {
    __shared__ float red[256];
    int t = threadIdx.x;
    red[t] = v; __syncthreads();
    for (int off = 128; off > 0; off >>= 1) {
        if (t < off) red[t] += red[t + off];
        __syncthreads();
    }
    float r = red[0]; __syncthreads();
    return r;
}
__device__ inline float block_reduce_max256(float v) {
    __shared__ float red[256];
    int t = threadIdx.x;
    red[t] = v; __syncthreads();
    for (int off = 128; off > 0; off >>= 1) {
        if (t < off) red[t] = fmaxf(red[t], red[t + off]);
        __syncthreads();
    }
    float r = red[0]; __syncthreads();
    return r;
}
// monotonic float<->uint encoding for atomic max (memset-0 is identity)
__device__ inline unsigned fenc(float x) {
    unsigned u = __float_as_uint(x);
    return (u & 0x80000000u) ? ~u : (u | 0x80000000u);
}
__device__ inline float fdec(unsigned u) {
    unsigned b = (u & 0x80000000u) ? (u & 0x7fffffffu) : ~u;
    return __uint_as_float(b);
}

// ------------------------------ WMMA GEMM ----------------------------------
// Each wave32 computes one 16x16 f32 tile via v_wmma_f32_16x16x32_f16.
// Block = 128 threads = 4 waves stacked along M (64 rows / block).
// A: [M,K] f16 (TRANSA -> A[k*lda+m]).  B: TRANSB -> B[n*ldb+k] (weights are
// staged transposed so k-pairs are contiguous).  When STAGEB, the block's
// 16xK B strip is DMA'd to LDS in 64-wide K chunks by the TDM and all 4
// waves read the fragment from LDS.  C mapping follows the ISA C/D layout
// (VGPR i: lanes 0-15 M=i, lanes 16-31 M=i+8).
template<bool TRANSA, bool TRANSB, bool GELU, bool ATOMIC, bool STAGEB>
__global__ __launch_bounds__(128) void gemm_kernel(
    const _Float16* __restrict__ Aall, long long aBatchStride, int lda,
    const _Float16* __restrict__ Ball, long long bSelStride, int ldb, int nb,
    float* __restrict__ outF, _Float16* __restrict__ outH,
    long long cBatchStride, int ldc,
    const float* __restrict__ bias, long long biasSelStride,
    const float* __restrict__ residual, long long resBatchStride,
    const float* __restrict__ rowScale, long long rsBatchStride,
    int M, int N, int K, int ksplit)
{
    const int lane = threadIdx.x & 31;
    const int wave = threadIdx.x >> 5;
    const int half = lane >> 4;        // which 16-lane half of the wave
    const int lr   = lane & 15;

    const int bat  = blockIdx.z / ksplit;
    const int kch  = blockIdx.z % ksplit;
    const int Keff = K / ksplit;
    const int widx = (nb > 0) ? (bat % nb) : bat;

    const int mbase = blockIdx.y * 64 + wave * 16;
    const int nbase = blockIdx.x * 16;

    const _Float16* A = Aall + (long long)bat * aBatchStride
        + (TRANSA ? (long long)(kch * Keff) * lda : (long long)(kch * Keff));
    const _Float16* Bm = Ball + (long long)widx * bSelStride
        + (TRANSB ? (long long)(kch * Keff) : (long long)(kch * Keff) * ldb);

    union { v8f v; float f[8]; } acc;
#pragma unroll
    for (int i = 0; i < 8; ++i) acc.f[i] = 0.0f;

    constexpr bool kTDM = STAGEB && TRANSB && (HAVE_TDM != 0);
    const int KSTEP = kTDM ? 64 : 32;
    __shared__ _Float16 btile[16 * 64];   // 2KB; unused (elided) when !kTDM

    for (int k0 = 0; k0 < Keff; k0 += KSTEP) {
#if HAVE_TDM
        if (kTDM) {
            __syncthreads();              // previous chunk fully consumed
            if (threadIdx.x < 32) {       // one wave issues the DMA + waits
                tdm_load_b_tile((unsigned)(unsigned long long)(size_t)btile,
                                Bm + (long long)nbase * ldb + k0, 64, 16, ldb);
                __builtin_amdgcn_s_wait_tensorcnt(0);
            }
            __syncthreads();
        }
#endif
        if (!TRANSA && k0 + KSTEP < Keff)   // prefetch next A chunk
            __builtin_prefetch(A + (long long)(mbase + lr) * lda + k0 + KSTEP, 0, 0);

        for (int kc = 0; kc < KSTEP; kc += 32) {
            union { v16h v; _Float16 h[16]; } a, b;
            // A 16x32 layout: lanes 0-15 K={0..7,16..23}, lanes 16-31 K={8..15,24..31}
#pragma unroll
            for (int i = 0; i < 8; ++i) {
                int kk = k0 + kc +
                         ((i < 4) ? (half * 8 + 2 * i) : (16 + half * 8 + 2 * (i - 4)));
                if (TRANSA) {
                    a.h[2 * i]     = A[(long long)kk * lda + (mbase + lr)];
                    a.h[2 * i + 1] = A[(long long)(kk + 1) * lda + (mbase + lr)];
                } else {
                    a.h[2 * i]     = A[(long long)(mbase + lr) * lda + kk];
                    a.h[2 * i + 1] = A[(long long)(mbase + lr) * lda + kk + 1];
                }
            }
            // B 32x16 layout: lanes 0-15 K={0..7,16..23}, lanes 16-31 K={8..15,24..31} (col = lr)
#pragma unroll
            for (int i = 0; i < 8; ++i) {
                int kl = kc + ((i < 4) ? (half * 8 + 2 * i) : (16 + half * 8 + 2 * (i - 4)));
                if (kTDM) {
                    b.h[2 * i]     = btile[lr * 64 + kl];
                    b.h[2 * i + 1] = btile[lr * 64 + kl + 1];
                } else if (TRANSB) {
                    b.h[2 * i]     = Bm[(long long)(nbase + lr) * ldb + k0 + kl];
                    b.h[2 * i + 1] = Bm[(long long)(nbase + lr) * ldb + k0 + kl + 1];
                } else {
                    b.h[2 * i]     = Bm[(long long)(k0 + kl) * ldb + (nbase + lr)];
                    b.h[2 * i + 1] = Bm[(long long)(k0 + kl + 1) * ldb + (nbase + lr)];
                }
            }
            acc.v = __builtin_amdgcn_wmma_f32_16x16x32_f16(
                false, a.v, false, b.v, (short)0, acc.v, false, false);
        }
    }

    const int col = nbase + lr;
#pragma unroll
    for (int i = 0; i < 8; ++i) {
        const int row = mbase + half * 8 + i;
        const long long cidx = (long long)bat * cBatchStride + (long long)row * ldc + col;
        float vv = acc.f[i];
        if (ATOMIC) {
            atomicAdd(&outF[cidx], vv);
        } else {
            if (bias)     vv += bias[(long long)widx * biasSelStride + col];
            if (rowScale) vv *= rowScale[(long long)bat * rsBatchStride + row];
            if (residual) vv += residual[(long long)bat * resBatchStride + (long long)row * ldc + col];
            if (GELU)     vv = 0.5f * vv * (1.0f + erff(vv * 0.70710678118654752f));
            if (outF) outF[cidx] = vv;
            if (outH) outH[cidx] = (_Float16)vv;
        }
    }
}

// ----------------------- staging / conversion kernels ----------------------
__global__ void convT_kernel(const float* __restrict__ src, _Float16* __restrict__ dst,
                             int K, int N) {   // dst[n*K+k] = src[k*N+n], batched in z
    long long base = (long long)blockIdx.z * K * N;
    int idx = blockIdx.x * 256 + threadIdx.x;
    if (idx >= K * N) return;
    int kk = idx / N, n = idx % N;
    dst[base + (long long)n * K + kk] = (_Float16)src[base + idx];
}
__global__ void conv_kernel(const float* __restrict__ src, _Float16* __restrict__ dst,
                            long long n) {
    long long idx = (long long)blockIdx.x * 256 + threadIdx.x;
    if (idx < n) dst[idx] = (_Float16)src[idx];
}

// ------------------------------ LayerNorm ----------------------------------
__global__ void ln_kernel(const float* __restrict__ x, const float* __restrict__ gam,
                          const float* __restrict__ bet, int L, int nb,
                          float* __restrict__ outF, _Float16* __restrict__ outH) {
    int row = blockIdx.x, t = threadIdx.x;
    int widx = (row / L) % nb;
    float v = x[(long long)row * 256 + t];
    float mu = block_reduce_sum256(v) * (1.0f / 256.0f);
    float d = v - mu;
    float var = block_reduce_sum256(d * d) * (1.0f / 256.0f);
    float y = d * rsqrtf(var + LN_EPS_) * gam[widx * 256 + t] + bet[widx * 256 + t];
    long long idx = (long long)row * 256 + t;
    if (outF) outF[idx] = y;
    if (outH) outH[idx] = (_Float16)y;
}

// ------------------------------ GCN pieces ---------------------------------
__global__ void deg_kernel(const int* __restrict__ col, float* __restrict__ deg) {
    int idx = blockIdx.x * 256 + threadIdx.x;
    if (idx < NEDGES_) atomicAdd(&deg[col[idx]], 1.0f);
}
__global__ void dinvdeg_kernel(float* __restrict__ deg) {
    int i = blockIdx.x * 256 + threadIdx.x;
    if (i < G_) { float d = deg[i]; deg[i] = d > 0.0f ? rsqrtf(d) : 0.0f; }
}
__global__ void edge_kernel(const int* __restrict__ row, const int* __restrict__ col,
                            const float* __restrict__ dv, const float* __restrict__ xw,
                            float* __restrict__ agg) {
    int e = blockIdx.x, t = threadIdx.x;
    int r = row[e], c = col[e];
    float w = dv[r] * dv[c];
    atomicAdd(&agg[((long long)c << 8) + t],            xw[((long long)r << 8) + t] * w);
    atomicAdd(&agg[(((long long)G_ + c) << 8) + t],     xw[(((long long)G_ + r) << 8) + t] * w);
}
__global__ void combine_kernel(const float* __restrict__ xln, const float* __restrict__ agg,
                               const float* __restrict__ bg, float* __restrict__ x1) {
    long long idx = (long long)blockIdx.x * 256 + threadIdx.x;
    x1[idx] = xln[idx] + agg[idx] + bg[idx & 255];
}

// ---------------------------- routing (sort) -------------------------------
__global__ void score_kernel(const float* __restrict__ x1, const float* __restrict__ Wr,
                             const float* __restrict__ br, float* __restrict__ sc) {
    int row = blockIdx.x, t = threadIdx.x;
    float p = x1[(long long)row * 256 + t] * Wr[t];
    float tot = block_reduce_sum256(p);
    if (t == 0) sc[row] = tot + br[0];
}
__global__ void sortinit_kernel(const float* __restrict__ sc, float* __restrict__ key,
                                int* __restrict__ val) {
    int idx = blockIdx.x * 256 + threadIdx.x;     // B_*SORTN
    int b = idx / SORTN, i = idx % SORTN;
    if (i < G_) { key[idx] = -sc[b * G_ + i]; val[idx] = i; }
    else        { key[idx] = INFINITY;        val[idx] = i; }
}
__global__ void bitonic_kernel(float* __restrict__ key, int* __restrict__ val, int j, int k) {
    int tid = blockIdx.x * 256 + threadIdx.x;
    int b = tid / SORTN, i = tid % SORTN;
    int ixj = i ^ j;
    if (ixj <= i) return;
    long long o = (long long)b * SORTN;
    float a = key[o + i], c = key[o + ixj];
    bool up = ((i & k) == 0);
    if (up ? (a > c) : (a < c)) {
        key[o + i] = c; key[o + ixj] = a;
        int t0 = val[o + i]; val[o + i] = val[o + ixj]; val[o + ixj] = t0;
    }
}
__global__ void gather_kernel(const float* __restrict__ x, const int* __restrict__ ord,
                              float* __restrict__ xg) {
    long long idx = (long long)blockIdx.x * 256 + threadIdx.x;
    long long tok = idx >> 8; int c = idx & 255;
    int b = (int)(tok / G_); int i = (int)(tok % G_);
    int src = ord[(long long)b * SORTN + i];
    xg[idx] = x[(((long long)b * G_ + src) << 8) + c];
}
__global__ void scatter_kernel(const float* __restrict__ yb, const int* __restrict__ ord,
                               float* __restrict__ xo) {
    long long idx = (long long)blockIdx.x * 256 + threadIdx.x;
    long long tok = idx >> 8; int c = idx & 255;
    int b = (int)(tok / G_); int i = (int)(tok % G_);
    int dst = ord[(long long)b * SORTN + i];
    xo[(((long long)b * G_ + dst) << 8) + c] = yb[idx];
}

// ----------------------- FAVOR+ softmax-kernel pieces ----------------------
__global__ void featprep_kernel(const float* __restrict__ q, _Float16* __restrict__ qs,
                                float* __restrict__ diag) {
    int row = blockIdx.x, t = threadIdx.x;
    long long idx = (long long)row * 256 + t;
    float xs = q[idx] * DN_;
    qs[idx] = (_Float16)xs;
    __shared__ float red[256];
    red[t] = xs * xs; __syncthreads();
    for (int off = 32; off > 0; off >>= 1) {
        if ((t & 63) < off) red[t] += red[t + off];
        __syncthreads();
    }
    if ((t & 63) == 0) diag[(long long)row * 4 + (t >> 6)] = 0.5f * red[t];
}
__global__ void vtrans_kernel(const float* __restrict__ v, _Float16* __restrict__ vt, int L) {
    long long idx = (long long)blockIdx.x * 256 + threadIdx.x;
    long long tok = idx >> 8; int c = idx & 255;
    int s = (int)(tok / L); int n = (int)(tok % L);
    vt[((long long)s * 256 + c) * L + n] = (_Float16)v[idx];
}
__global__ void kmax_kernel(const float* __restrict__ kd, unsigned* __restrict__ enc, int L) {
    int s = blockIdx.y;
    const float* base = kd + (long long)s * L * 256;
    long long total = (long long)L * 256;
    float m = -INFINITY;
    for (long long i = blockIdx.x * 256 + threadIdx.x; i < total;
         i += (long long)gridDim.x * 256)
        m = fmaxf(m, base[i]);
    m = block_reduce_max256(m);
    if (threadIdx.x == 0) atomicMax(&enc[s], fenc(m));
}
__global__ void qfeat_kernel(float* __restrict__ qd, _Float16* __restrict__ qp16,
                             const float* __restrict__ diag, int h) {
    int row = blockIdx.x, t = threadIdx.x;
    long long idx = (long long)row * 256 + t;
    float v = qd[idx];
    float mx = block_reduce_max256(v);
    float e = RATIO_ * (expf(v - diag[(long long)row * 4 + h] - mx) + KEPS_);
    qd[idx] = e; qp16[idx] = (_Float16)e;
}
__global__ void kfeat_kernel(float* __restrict__ kd, _Float16* __restrict__ kp16,
                             const float* __restrict__ diag, const unsigned* __restrict__ enc,
                             int h, int L) {
    long long idx = (long long)blockIdx.x * 256 + threadIdx.x;
    int row = (int)(idx >> 8); int s = row / L;
    float mx = fdec(enc[s]);
    float e = RATIO_ * (expf(kd[idx] - diag[(long long)row * 4 + h] - mx) + KEPS_);
    kd[idx] = e; kp16[idx] = (_Float16)e;
}
__global__ void ksum_kernel(const float* __restrict__ kp, float* __restrict__ ks, int L) {
    int s = blockIdx.y, c = blockIdx.x, t = threadIdx.x;
    int per = (L + gridDim.x - 1) / gridDim.x;
    int n0 = c * per, n1 = n0 + per; if (n1 > L) n1 = L;
    float acc = 0.0f;
    for (int n = n0; n < n1; ++n) acc += kp[((long long)s * L + n) * 256 + t];
    atomicAdd(&ks[s * 256 + t], acc);
}
__global__ void dinv_kernel(const float* __restrict__ qp, const float* __restrict__ ks,
                            float* __restrict__ dv, int L) {
    int row = blockIdx.x, t = threadIdx.x;
    int s = row / L;
    float p = qp[(long long)row * 256 + t] * ks[s * 256 + t];
    float tot = block_reduce_sum256(p);
    if (t == 0) dv[row] = 1.0f / tot;
}
__global__ void ctxcvt_kernel(const float* __restrict__ ctx, _Float16* __restrict__ ct, int S) {
    int idx = blockIdx.x * 256 + threadIdx.x;
    if (idx >= S * 256 * 64) return;
    int s = idx / (256 * 64); int r = idx % (256 * 64);
    int m = r / 64; int d = r % 64;
    ct[((long long)s * 64 + d) * 256 + m] = (_Float16)ctx[idx];   // [S][64][256]
}

// ---------------------------- host-side driver -----------------------------
struct PerfW {
    const float *ln1_g, *ln1_b, *bq, *bk, *bv, *bo, *ln2_g, *ln2_b, *b1, *b2;
    const _Float16 *wq, *wk, *wv, *wo, *proj, *w1, *w2;
    int nb;
};
struct Scr {
    float *q, *k, *v_o, *qdf, *kdf, *x2;
    _Float16 *h_y, *h_qs, *h_ks, *h_vt, *h_qp, *h_kp, *h_o, *h_ff;
    float *diagq, *diagk, *ksum, *dinvb, *ctx;
    unsigned *kmaxe;
    _Float16 *ctxt;
};

static void performer_pass(hipStream_t st, const float* xin, float* xout,
                           int S, int L, const PerfW& W, const Scr& R)
{
    const dim3 blk(128);
    const long long tok256 = (long long)L * 256;

    // y = LN(x)
    ln_kernel<<<NTOK, 256, 0, st>>>(xin, W.ln1_g, W.ln1_b, L, W.nb, nullptr, R.h_y);
    // q,k,v = y @ W + b
    dim3 gqkv(E_ / 16, L / 64, S);
    gemm_kernel<false, true, false, false, true><<<gqkv, blk, 0, st>>>(
        R.h_y, tok256, 256, W.wq, (long long)E_ * E_, 256, W.nb,
        R.q, nullptr, tok256, 256, W.bq, 256, nullptr, 0, nullptr, 0, L, E_, E_, 1);
    gemm_kernel<false, true, false, false, true><<<gqkv, blk, 0, st>>>(
        R.h_y, tok256, 256, W.wk, (long long)E_ * E_, 256, W.nb,
        R.k, nullptr, tok256, 256, W.bk, 256, nullptr, 0, nullptr, 0, L, E_, E_, 1);
    gemm_kernel<false, true, false, false, true><<<gqkv, blk, 0, st>>>(
        R.h_y, tok256, 256, W.wv, (long long)E_ * E_, 256, W.nb,
        R.v_o, nullptr, tok256, 256, W.bv, 256, nullptr, 0, nullptr, 0, L, E_, E_, 1);
    // scaled f16 copies + per-head diag; v transposed [s][c][n] for ctx GEMM
    featprep_kernel<<<NTOK, 256, 0, st>>>(R.q, R.h_qs, R.diagq);
    featprep_kernel<<<NTOK, 256, 0, st>>>(R.k, R.h_ks, R.diagk);
    vtrans_kernel<<<NTOK, 256, 0, st>>>(R.v_o, R.h_vt, L);

    const int ksplit = L / 160;                 // Keff = 160, multiple of 32
    for (int h = 0; h < HEADS_; ++h) {
        // qd/kd = xs @ proj^T   (K = 64)
        dim3 gfeat(MFEAT_ / 16, L / 64, S);
        gemm_kernel<false, true, false, false, true><<<gfeat, blk, 0, st>>>(
            R.h_qs + h * DH_, tok256, 256, W.proj, (long long)MFEAT_ * DH_, DH_, W.nb,
            R.qdf, nullptr, tok256, 256, nullptr, 0, nullptr, 0, nullptr, 0, L, MFEAT_, DH_, 1);
        gemm_kernel<false, true, false, false, true><<<gfeat, blk, 0, st>>>(
            R.h_ks + h * DH_, tok256, 256, W.proj, (long long)MFEAT_ * DH_, DH_, W.nb,
            R.kdf, nullptr, tok256, 256, nullptr, 0, nullptr, 0, nullptr, 0, L, MFEAT_, DH_, 1);
        // k global max per sequence, then positive features
        hipMemsetAsync(R.kmaxe, 0, S * sizeof(unsigned), st);
        kmax_kernel<<<dim3(64, S), 256, 0, st>>>(R.kdf, R.kmaxe, L);
        qfeat_kernel<<<NTOK, 256, 0, st>>>(R.qdf, R.h_qp, R.diagq, h);
        kfeat_kernel<<<NTOK, 256, 0, st>>>(R.kdf, R.h_kp, R.diagk, R.kmaxe, h, L);
        // k_sum and d_inv
        hipMemsetAsync(R.ksum, 0, (long long)S * 256 * sizeof(float), st);
        ksum_kernel<<<dim3(64, S), 256, 0, st>>>(R.kdf, R.ksum, L);
        dinv_kernel<<<NTOK, 256, 0, st>>>(R.qdf, R.ksum, R.dinvb, L);
        // ctx[m,d] = sum_n kp[n,m] * v[n,d]  (split-K, f32 atomic accumulate)
        hipMemsetAsync(R.ctx, 0, (long long)S * 256 * 64 * sizeof(float), st);
        gemm_kernel<true, true, false, true, false>
            <<<dim3(DH_ / 16, MFEAT_ / 64, S * ksplit), blk, 0, st>>>(
            R.h_kp, tok256, 256, R.h_vt + (long long)h * DH_ * L, (long long)256 * L, L, 0,
            R.ctx, nullptr, (long long)256 * 64, 64, nullptr, 0, nullptr, 0, nullptr, 0,
            MFEAT_, DH_, L, ksplit);
        ctxcvt_kernel<<<(S * 256 * 64 + 255) / 256, 256, 0, st>>>(R.ctx, R.ctxt, S);
        // o[:, h*64:+64] = (qp @ ctx) * dinv[n]   (also emit f16)
        gemm_kernel<false, true, false, false, true>
            <<<dim3(DH_ / 16, L / 64, S), blk, 0, st>>>(
            R.h_qp, tok256, 256, R.ctxt, (long long)64 * 256, 256, 0,
            R.v_o + h * DH_, R.h_o + h * DH_, tok256, 256,
            nullptr, 0, nullptr, 0, R.dinvb, L, L, DH_, MFEAT_, 1);
    }
    // x2 = x + o @ Wo + bo
    gemm_kernel<false, true, false, false, true><<<gqkv, blk, 0, st>>>(
        R.h_o, tok256, 256, W.wo, (long long)E_ * E_, 256, W.nb,
        R.x2, nullptr, tok256, 256, W.bo, 256, xin, tok256, nullptr, 0, L, E_, E_, 1);
    // FF: gelu(LN(x2) @ W1 + b1) @ W2 + b2 + x2
    ln_kernel<<<NTOK, 256, 0, st>>>(R.x2, W.ln2_g, W.ln2_b, L, W.nb, nullptr, R.h_y);
    gemm_kernel<false, true, true, false, true>
        <<<dim3(DFF_ / 16, L / 64, S), blk, 0, st>>>(
        R.h_y, tok256, 256, W.w1, (long long)E_ * DFF_, 256, W.nb,
        nullptr, R.h_ff, (long long)L * DFF_, DFF_, W.b1, DFF_, nullptr, 0, nullptr, 0,
        L, DFF_, E_, 1);
    gemm_kernel<false, true, false, false, true><<<gqkv, blk, 0, st>>>(
        R.h_ff, (long long)L * DFF_, DFF_, W.w2, (long long)DFF_ * E_, DFF_, W.nb,
        xout, nullptr, tok256, 256, W.b2, 256, R.x2, tok256, nullptr, 0, L, E_, DFF_, 1);
}

extern "C" void kernel_launch(void* const* d_in, const int* in_sizes, int n_in,
                              void* d_out, int out_size, void* d_ws, size_t ws_size,
                              hipStream_t stream)
{
    (void)in_sizes; (void)n_in; (void)out_size; (void)ws_size;
    hipStream_t st = stream;

    // ------------------------- inputs --------------------------------------
    const float* x0    = (const float*)d_in[0];
    const int*   graph = (const int*)d_in[1];
    const int*   erow  = graph;
    const int*   ecol  = graph + NEDGES_;
    const float* ln0_g = (const float*)d_in[2];
    const float* ln0_b = (const float*)d_in[3];
    const float* Wg    = (const float*)d_in[4];
    const float* bg    = (const float*)d_in[5];
    const float* Wr    = (const float*)d_in[6];
    const float* br    = (const float*)d_in[7];
    // pb (prefix [BINS]): 8..24 ; pf: 25..41  (dict insertion order)
    const float* pb_ln1g = (const float*)d_in[8];
    const float* pb_ln1b = (const float*)d_in[9];
    const float* pb_Wq   = (const float*)d_in[10];
    const float* pb_bq   = (const float*)d_in[11];
    const float* pb_Wk   = (const float*)d_in[12];
    const float* pb_bk   = (const float*)d_in[13];
    const float* pb_Wv   = (const float*)d_in[14];
    const float* pb_bv   = (const float*)d_in[15];
    const float* pb_Wo   = (const float*)d_in[16];
    const float* pb_bo   = (const float*)d_in[17];
    const float* pb_proj = (const float*)d_in[18];
    const float* pb_ln2g = (const float*)d_in[19];
    const float* pb_ln2b = (const float*)d_in[20];
    const float* pb_W1   = (const float*)d_in[21];
    const float* pb_b1   = (const float*)d_in[22];
    const float* pb_W2   = (const float*)d_in[23];
    const float* pb_b2   = (const float*)d_in[24];
    const float* pf_ln1g = (const float*)d_in[25];
    const float* pf_ln1b = (const float*)d_in[26];
    const float* pf_Wq   = (const float*)d_in[27];
    const float* pf_bq   = (const float*)d_in[28];
    const float* pf_Wk   = (const float*)d_in[29];
    const float* pf_bk   = (const float*)d_in[30];
    const float* pf_Wv   = (const float*)d_in[31];
    const float* pf_bv   = (const float*)d_in[32];
    const float* pf_Wo   = (const float*)d_in[33];
    const float* pf_bo   = (const float*)d_in[34];
    const float* pf_proj = (const float*)d_in[35];
    const float* pf_ln2g = (const float*)d_in[36];
    const float* pf_ln2b = (const float*)d_in[37];
    const float* pf_W1   = (const float*)d_in[38];
    const float* pf_b1   = (const float*)d_in[39];
    const float* pf_W2   = (const float*)d_in[40];
    const float* pf_b2   = (const float*)d_in[41];
    float* out = (float*)d_out;

    // ------------------------- workspace carve -----------------------------
    char* wp = (char*)d_ws;
    auto alloc = [&](long long bytes) -> void* {
        void* r = (void*)wp;
        wp += (bytes + 255) & ~(long long)255;
        return r;
    };
    const long long T256 = (long long)NTOK * 256;
    // f16 staged weights (transposed: [out][in])
    _Float16* wg16     = (_Float16*)alloc(2LL * 65536);
    _Float16* pbwq16   = (_Float16*)alloc(2LL * BINS_ * 65536);
    _Float16* pbwk16   = (_Float16*)alloc(2LL * BINS_ * 65536);
    _Float16* pbwv16   = (_Float16*)alloc(2LL * BINS_ * 65536);
    _Float16* pbwo16   = (_Float16*)alloc(2LL * BINS_ * 65536);
    _Float16* pbproj16 = (_Float16*)alloc(2LL * BINS_ * MFEAT_ * DH_);
    _Float16* pbw116   = (_Float16*)alloc(2LL * BINS_ * E_ * DFF_);
    _Float16* pbw216   = (_Float16*)alloc(2LL * BINS_ * E_ * DFF_);
    _Float16* pfwq16   = (_Float16*)alloc(2LL * 65536);
    _Float16* pfwk16   = (_Float16*)alloc(2LL * 65536);
    _Float16* pfwv16   = (_Float16*)alloc(2LL * 65536);
    _Float16* pfwo16   = (_Float16*)alloc(2LL * 65536);
    _Float16* pfproj16 = (_Float16*)alloc(2LL * MFEAT_ * DH_);
    _Float16* pfw116   = (_Float16*)alloc(2LL * E_ * DFF_);
    _Float16* pfw216   = (_Float16*)alloc(2LL * E_ * DFF_);
    // f32 activation buffers
    float* bufA = (float*)alloc(4 * T256);  // xln / xg
    float* bufB = (float*)alloc(4 * T256);  // xw / x1 / yb
    float* bufC = (float*)alloc(4 * T256);  // agg / q
    float* bufD = (float*)alloc(4 * T256);  // k
    float* bufE = (float*)alloc(4 * T256);  // v / o
    float* bufF = (float*)alloc(4 * T256);  // qd / qp
    float* bufG = (float*)alloc(4 * T256);  // kd / kp
    float* bufH = (float*)alloc(4 * T256);  // x2
    float* bufI = (float*)alloc(4 * T256);  // scattered-back x
    // f16 activation buffers
    _Float16* h_y  = (_Float16*)alloc(2 * T256);
    _Float16* h_qs = (_Float16*)alloc(2 * T256);
    _Float16* h_ks = (_Float16*)alloc(2 * T256);
    _Float16* h_vt = (_Float16*)alloc(2 * T256);
    _Float16* h_qp = (_Float16*)alloc(2 * T256);
    _Float16* h_kp = (_Float16*)alloc(2 * T256);
    _Float16* h_o  = (_Float16*)alloc(2 * T256);
    _Float16* h_ff = (_Float16*)alloc(2LL * NTOK * DFF_);
    // small
    float*    deg    = (float*)alloc(4LL * G_);
    float*    scores = (float*)alloc(4LL * NTOK);
    float*    skey   = (float*)alloc(4LL * B_ * SORTN);
    int*      sval   = (int*)alloc(4LL * B_ * SORTN);
    float*    diagq  = (float*)alloc(4LL * NTOK * 4);
    float*    diagk  = (float*)alloc(4LL * NTOK * 4);
    float*    ksum   = (float*)alloc(4LL * SMAX * 256);
    float*    dinvb  = (float*)alloc(4LL * NTOK);
    float*    ctx    = (float*)alloc(4LL * SMAX * 256 * 64);
    _Float16* ctxt   = (_Float16*)alloc(2LL * SMAX * 64 * 256);
    unsigned* kmaxe  = (unsigned*)alloc(4LL * SMAX);

    // ------------------------- stage weights to f16 ------------------------
    auto convT = [&](const float* s, _Float16* d, int K, int N, int nb) {
        convT_kernel<<<dim3((K * N + 255) / 256, 1, nb), 256, 0, st>>>(s, d, K, N);
    };
    convT(Wg, wg16, 256, 256, 1);
    convT(pb_Wq, pbwq16, 256, 256, BINS_);  convT(pb_Wk, pbwk16, 256, 256, BINS_);
    convT(pb_Wv, pbwv16, 256, 256, BINS_);  convT(pb_Wo, pbwo16, 256, 256, BINS_);
    convT(pb_W1, pbw116, 256, 1024, BINS_); convT(pb_W2, pbw216, 1024, 256, BINS_);
    conv_kernel<<<(BINS_ * MFEAT_ * DH_ + 255) / 256, 256, 0, st>>>(
        pb_proj, pbproj16, (long long)BINS_ * MFEAT_ * DH_);
    convT(pf_Wq, pfwq16, 256, 256, 1);  convT(pf_Wk, pfwk16, 256, 256, 1);
    convT(pf_Wv, pfwv16, 256, 256, 1);  convT(pf_Wo, pfwo16, 256, 256, 1);
    convT(pf_W1, pfw116, 256, 1024, 1); convT(pf_W2, pfw216, 1024, 256, 1);
    conv_kernel<<<(MFEAT_ * DH_ + 255) / 256, 256, 0, st>>>(
        pf_proj, pfproj16, (long long)MFEAT_ * DH_);

    // ------------------------- ln0 + GCNConv -------------------------------
    ln_kernel<<<NTOK, 256, 0, st>>>(x0, ln0_g, ln0_b, NTOK, 1, bufA, h_y);
    // xw = xln @ Wg  (one batch, M = NTOK)
    gemm_kernel<false, true, false, false, true>
        <<<dim3(E_ / 16, NTOK / 64, 1), dim3(128), 0, st>>>(
        h_y, 0, 256, wg16, 0, 256, 1, bufB, nullptr, 0, 256,
        nullptr, 0, nullptr, 0, nullptr, 0, NTOK, E_, E_, 1);
    hipMemsetAsync(deg, 0, 4LL * G_, st);
    deg_kernel<<<(NEDGES_ + 255) / 256, 256, 0, st>>>(ecol, deg);
    dinvdeg_kernel<<<(G_ + 255) / 256, 256, 0, st>>>(deg);
    hipMemsetAsync(bufC, 0, 4 * T256, st);
    edge_kernel<<<NEDGES_, 256, 0, st>>>(erow, ecol, deg, bufB, bufC);
    combine_kernel<<<NTOK, 256, 0, st>>>(bufA, bufC, bg, bufB);   // x1

    // ------------------------- routing: sort + gather ----------------------
    score_kernel<<<NTOK, 256, 0, st>>>(bufB, Wr, br, scores);
    sortinit_kernel<<<(B_ * SORTN) / 256, 256, 0, st>>>(scores, skey, sval);
    for (int k = 2; k <= SORTN; k <<= 1)
        for (int j = k >> 1; j > 0; j >>= 1)
            bitonic_kernel<<<(B_ * SORTN) / 256, 256, 0, st>>>(skey, sval, j, k);
    gather_kernel<<<NTOK, 256, 0, st>>>(bufB, sval, bufA);        // xg

    // ------------------------- performer passes ----------------------------
    Scr R{bufC, bufD, bufE, bufF, bufG, bufH,
          h_y, h_qs, h_ks, h_vt, h_qp, h_kp, h_o, h_ff,
          diagq, diagk, ksum, dinvb, ctx, kmaxe, ctxt};
    PerfW wb; wb.nb = BINS_;
    wb.ln1_g = pb_ln1g; wb.ln1_b = pb_ln1b; wb.bq = pb_bq; wb.bk = pb_bk; wb.bv = pb_bv;
    wb.bo = pb_bo; wb.ln2_g = pb_ln2g; wb.ln2_b = pb_ln2b; wb.b1 = pb_b1; wb.b2 = pb_b2;
    wb.wq = pbwq16; wb.wk = pbwk16; wb.wv = pbwv16; wb.wo = pbwo16;
    wb.proj = pbproj16; wb.w1 = pbw116; wb.w2 = pbw216;
    PerfW wf; wf.nb = 1;
    wf.ln1_g = pf_ln1g; wf.ln1_b = pf_ln1b; wf.bq = pf_bq; wf.bk = pf_bk; wf.bv = pf_bv;
    wf.bo = pf_bo; wf.ln2_g = pf_ln2g; wf.ln2_b = pf_ln2b; wf.b1 = pf_b1; wf.b2 = pf_b2;
    wf.wq = pfwq16; wf.wk = pfwk16; wf.wv = pfwv16; wf.wo = pfwo16;
    wf.proj = pfproj16; wf.w1 = pfw116; wf.w2 = pfw216;

    // per-bin performers: 20 sequences of 1920 tokens (weights selected bin = s % 10)
    performer_pass(st, bufA, bufB, B_ * BINS_, LBIN, wb, R);
    // scatter back to original order
    scatter_kernel<<<NTOK, 256, 0, st>>>(bufB, sval, bufI);
    // full-sequence performer: 2 sequences of 19200 tokens, write d_out
    performer_pass(st, bufI, out, B_, G_, wf, R);
}
